// MultiHeadedSelfAttention_16638703305486
// MI455X (gfx1250) — matrix operand order repaired and checked by reading, not compiled, not run on hardware
//
#include <hip/hip_runtime.h>

typedef unsigned short u16;
typedef unsigned int   u32;
typedef __attribute__((ext_vector_type(16))) __bf16 v16bf;
typedef __attribute__((ext_vector_type(2)))  __bf16 v2bf;
typedef __attribute__((ext_vector_type(8)))  float  v8f;

union V16 {
  v16bf v;
  u32   w[8];
  u16   u[16];
  __device__ V16() {}
};

// hardware bf16 conversion (RNE) -> raw bits
__device__ inline u16 bfbits(float f) {
  __bf16 h = (__bf16)f;
  return __builtin_bit_cast(u16, h);
}
// pack two f32 -> packed bf16x2 dword (v_cvt_pk_bf16_f32)
__device__ inline u32 pk2bf(float a, float b) {
  v2bf t; t[0] = (__bf16)a; t[1] = (__bf16)b;
  return __builtin_bit_cast(u32, t);
}

__device__ inline v8f vzero() {
  v8f z = {0.f, 0.f, 0.f, 0.f, 0.f, 0.f, 0.f, 0.f};
  return z;
}

// load 16 bf16 from two 16B-aligned 8-element chunks
__device__ inline void ld16(V16& d, const u16* lo, const u16* hi) {
  uint4 a = *reinterpret_cast<const uint4*>(lo);
  uint4 b = *reinterpret_cast<const uint4*>(hi);
  d.w[0] = a.x; d.w[1] = a.y; d.w[2] = a.z; d.w[3] = a.w;
  d.w[4] = b.x; d.w[5] = b.y; d.w[6] = b.z; d.w[7] = b.w;
}

// async global->LDS DMA of 16 bytes (ASYNCcnt-tracked, no VGPR round trip)
__device__ inline void async_ld128(u32 lds_byte_off, const u16* gp) {
  asm volatile("global_load_async_to_lds_b128 %0, %1, off"
               :: "v"(lds_byte_off), "v"(gp) : "memory");
}
__device__ inline u32 ldsoff(const u16* p) {  // generic ptr -> LDS byte offset
  return (u32)(size_t)p;
}

__device__ inline float half_max(float t) {
  t = fmaxf(t, __shfl_xor(t, 1));
  t = fmaxf(t, __shfl_xor(t, 2));
  t = fmaxf(t, __shfl_xor(t, 4));
  t = fmaxf(t, __shfl_xor(t, 8));
  return t;
}
__device__ inline float half_sum(float t) {
  t += __shfl_xor(t, 1);
  t += __shfl_xor(t, 2);
  t += __shfl_xor(t, 4);
  t += __shfl_xor(t, 8);
  return t;
}

#define LDX 40  // u16 row stride (32 data + 8 pad), keeps 16B alignment

// -------- one-shot converts (paid once; everything downstream is bf16) ----
__global__ __launch_bounds__(256) void cvt_x_kernel(
    const float* __restrict__ in, u16* __restrict__ out) {
  int i = (blockIdx.x * 256 + threadIdx.x) * 8;
  float4 a = *(const float4*)(in + i);
  float4 b = *(const float4*)(in + i + 4);
  uint4 r;
  r.x = pk2bf(a.x, a.y); r.y = pk2bf(a.z, a.w);
  r.z = pk2bf(b.x, b.y); r.w = pk2bf(b.z, b.w);
  *(uint4*)(out + i) = r;
}

// Wt[n][k] = bf16(W[k][n]); thread writes 8 consecutive k for one n
__global__ __launch_bounds__(256) void cvt_wt_kernel(
    const float* __restrict__ W, u16* __restrict__ Wt) {
  int idx = blockIdx.x * 256 + threadIdx.x;  // 32768 threads
  int n = idx >> 6, k0 = (idx & 63) * 8;
  uint4 r;
  u32* rw = (u32*)&r;
#pragma unroll
  for (int j = 0; j < 4; ++j)
    rw[j] = pk2bf(W[(k0 + 2 * j) * 512 + n], W[(k0 + 2 * j + 1) * 512 + n]);
  *(uint4*)(Wt + n * 512 + k0) = r;
}

// -------- GEMM: Y[4096,512] = X[4096,512] @ W[512,512] + bias --------------
// X bf16 row-major, Wt bf16 pre-transposed [n][k].
// Block tile 128(M) x 64(N); 8 waves, each owns a 2x2 grid of 16x16 subtiles.
// Double-buffered LDS filled by global_load_async_to_lds_b128.
// mode 0: scatter bf16 to [B,H,S,64] (Q or K)
// mode 1: scatter bf16 to [B,H,64,S] (V transposed)
// mode 2: f32 row-major to outf
__global__ __launch_bounds__(256) void gemm512_kernel(
    const u16* __restrict__ Xb, const u16* __restrict__ Wt,
    const float* __restrict__ bias,
    u16* __restrict__ outb, float* __restrict__ outf, int mode)
{
  __shared__ u16 XL[2][128 * LDX];
  __shared__ u16 WL[2][64 * LDX];  // WL[n][k]

  const int tid  = threadIdx.x;
  const int lane = tid & 31;
  const int wv   = tid >> 5;
  const int mb   = blockIdx.x * 128;
  const int nb   = blockIdx.y * 64;
  const int ms0  = (wv & 3) * 2;   // first of 2 m-subtiles
  const int ns0  = (wv >> 2) * 2;  // first of 2 n-subtiles
  const int m    = lane & 15;
  const int kh   = lane >> 4;

  // staging coords: X is 128x32 (512 chunks of 8), W is 64x32 (256 chunks)
  const int soff = (tid & 3) * 8;
  const int xr0  = tid >> 2;  // X rows 0..63 (second chunk handles +64)
  const u16* xg0 = Xb + (mb + xr0) * 512 + soff;
  const u16* xg1 = xg0 + 64 * 512;
  const u16* wg  = Wt + (nb + xr0) * 512 + soff;
  u32 xl0[2], xl1[2], wl[2];
#pragma unroll
  for (int b = 0; b < 2; ++b) {
    xl0[b] = ldsoff(&XL[b][xr0 * LDX + soff]);
    xl1[b] = ldsoff(&XL[b][(xr0 + 64) * LDX + soff]);
    wl[b]  = ldsoff(&WL[b][xr0 * LDX + soff]);
  }

  v8f acc00 = vzero(), acc01 = vzero(), acc10 = vzero(), acc11 = vzero();

  // prologue: stage k-slab 0 into buffer 0
  async_ld128(xl0[0], xg0); async_ld128(xl1[0], xg1); async_ld128(wl[0], wg);
  xg0 += 32; xg1 += 32; wg += 32;

  for (int i = 0; i < 16; ++i) {
    const int cur = i & 1, nxt = cur ^ 1;
    if (i < 15) {  // stage next slab into other buffer, then wait current
      async_ld128(xl0[nxt], xg0); async_ld128(xl1[nxt], xg1);
      async_ld128(wl[nxt], wg);
      xg0 += 32; xg1 += 32; wg += 32;
      asm volatile("s_wait_asynccnt 0x3" ::: "memory");
    } else {
      asm volatile("s_wait_asynccnt 0x0" ::: "memory");
    }
    __syncthreads();

    V16 a0, a1, b0, b1;
    {
      const u16* p = &XL[cur][(ms0 * 16 + m) * LDX];
      ld16(a0, p + 8 * kh, p + 16 + 8 * kh);
      p += 16 * LDX;
      ld16(a1, p + 8 * kh, p + 16 + 8 * kh);
      const u16* q = &WL[cur][(ns0 * 16 + m) * LDX];
      ld16(b0, q + 8 * kh, q + 16 + 8 * kh);
      q += 16 * LDX;
      ld16(b1, q + 8 * kh, q + 16 + 8 * kh);
    }
    acc00 = __builtin_amdgcn_wmma_f32_16x16x32_bf16(
        false, a0.v, false, b0.v, (short)0, acc00, false, false);
    acc01 = __builtin_amdgcn_wmma_f32_16x16x32_bf16(
        false, a0.v, false, b1.v, (short)0, acc01, false, false);
    acc10 = __builtin_amdgcn_wmma_f32_16x16x32_bf16(
        false, a1.v, false, b0.v, (short)0, acc10, false, false);
    acc11 = __builtin_amdgcn_wmma_f32_16x16x32_bf16(
        false, a1.v, false, b1.v, (short)0, acc11, false, false);
    __syncthreads();
  }

  // ---- epilogue: bias + scatter ----
  const int hl = lane >> 4;
#pragma unroll
  for (int ti = 0; ti < 2; ++ti) {
#pragma unroll
    for (int tj = 0; tj < 2; ++tj) {
      v8f a = ti ? (tj ? acc11 : acc10) : (tj ? acc01 : acc00);
      int col = nb + (ns0 + tj) * 16 + m;
      float bv = bias[col];
#pragma unroll
      for (int r = 0; r < 8; ++r) {
        float v = a[r] + bv;
        int rw = mb + (ms0 + ti) * 16 + r + 8 * hl;  // 0..4095 (= b*2048+s)
        if (mode == 2) {
          outf[rw * 512 + col] = v;
        } else {
          int bb = rw >> 11, s = rw & 2047;
          int h = col >> 6, d = col & 63;
          int idx = (mode == 0) ? (((bb * 8 + h) * 2048 + s) * 64 + d)
                                : (((bb * 8 + h) * 64 + d) * 2048 + s);
          outb[idx] = bfbits(v);
        }
      }
    }
  }
}

// -------- Fused flash attention: one wave per 16-query tile ----------------
// Q,K: [B,H,S,64] bf16   Vt: [B,H,64,S] bf16   ctx: [B,S,512] bf16
__global__ __launch_bounds__(256) void attn_kernel(
    const u16* __restrict__ Q, const u16* __restrict__ K,
    const u16* __restrict__ Vt, u16* __restrict__ ctx)
{
  __shared__ u16 P[8 * 16 * LDX];  // per-wave 16x32 probability tile

  const int tid  = threadIdx.x;
  const int lane = tid & 31;
  const int wv   = tid >> 5;
  const int bh   = blockIdx.y;               // b*8 + h
  const int qb   = (blockIdx.x * 8 + wv) * 16;
  const int m    = lane & 15;
  const int kh   = lane >> 4;
  const float C  = 0.125f * 1.44269504088896f;  // scale * log2(e)

  // Q A-fragments for both 32-wide d-slabs
  V16 qa0, qa1;
  {
    const u16* qp = Q + (bh * 2048 + qb + m) * 64;
    ld16(qa0, qp + 8 * kh,      qp + 16 + 8 * kh);
    ld16(qa1, qp + 32 + 8 * kh, qp + 48 + 8 * kh);
  }

  float mrun[8], srun[8];
  v8f acc[4];
#pragma unroll
  for (int r = 0; r < 8; ++r) { mrun[r] = -3.0e38f; srun[r] = 0.f; }
#pragma unroll
  for (int f = 0; f < 4; ++f) acc[f] = vzero();

  u16* Pw = &P[wv * 16 * LDX];
  const u16* Kb = K  + bh * 2048 * 64;
  const u16* Vb = Vt + bh * 64 * 2048;

  for (int kt = 0; kt < 64; ++kt) {
    const int kb = kt * 32;

    // ---- scores: two 16x16 D-frags over 32 keys, accumulate d-slabs ----
    v8f s0, s1;
    {
      V16 b;
      const u16* kp0 = Kb + (kb + m) * 64;        // keys kb..kb+15
      ld16(b, kp0 + 8 * kh, kp0 + 16 + 8 * kh);
      s0 = __builtin_amdgcn_wmma_f32_16x16x32_bf16(
          false, qa0.v, false, b.v, (short)0, vzero(), false, false);
      ld16(b, kp0 + 32 + 8 * kh, kp0 + 48 + 8 * kh);
      s0 = __builtin_amdgcn_wmma_f32_16x16x32_bf16(
          false, qa1.v, false, b.v, (short)0, s0, false, false);
      const u16* kp1 = Kb + (kb + 16 + m) * 64;   // keys kb+16..+31
      ld16(b, kp1 + 8 * kh, kp1 + 16 + 8 * kh);
      s1 = __builtin_amdgcn_wmma_f32_16x16x32_bf16(
          false, qa0.v, false, b.v, (short)0, vzero(), false, false);
      ld16(b, kp1 + 32 + 8 * kh, kp1 + 48 + 8 * kh);
      s1 = __builtin_amdgcn_wmma_f32_16x16x32_bf16(
          false, qa1.v, false, b.v, (short)0, s1, false, false);
    }

    // ---- online softmax (rows live in 16-lane halves) ----
    float p0[8], p1[8], corr[8];
#pragma unroll
    for (int r = 0; r < 8; ++r) {
      float tmax = half_max(fmaxf(s0[r], s1[r]));
      float mn = fmaxf(mrun[r], tmax);
      float e0 = exp2f((s0[r] - mn) * C);
      float e1 = exp2f((s1[r] - mn) * C);
      p0[r] = e0; p1[r] = e1;
      float rs = half_sum(e0 + e1);
      corr[r] = exp2f((mrun[r] - mn) * C);
      srun[r] = srun[r] * corr[r] + rs;
      mrun[r] = mn;
    }
#pragma unroll
    for (int f = 0; f < 4; ++f)
#pragma unroll
      for (int r = 0; r < 8; ++r) acc[f][r] *= corr[r];

    // ---- D-layout -> A-layout via wave-private LDS ----
#pragma unroll
    for (int r = 0; r < 8; ++r) {
      int row = r + 8 * kh;
      Pw[row * LDX + m]      = bfbits(p0[r]);
      Pw[row * LDX + 16 + m] = bfbits(p1[r]);
    }
    asm volatile("s_wait_dscnt 0" ::: "memory");
    V16 pa;
    {
      const u16* pr = Pw + m * LDX;
      ld16(pa, pr + 8 * kh, pr + 16 + 8 * kh);
    }

    // ---- context += P @ V (4 d-tiles of 16) ----
#pragma unroll
    for (int f = 0; f < 4; ++f) {
      V16 b;
      const u16* vp = Vb + (f * 16 + m) * 2048 + kb;
      ld16(b, vp + 8 * kh, vp + 16 + 8 * kh);
      acc[f] = __builtin_amdgcn_wmma_f32_16x16x32_bf16(
          false, pa.v, false, b.v, (short)0, acc[f], false, false);
    }
  }

  // ---- normalize and store ctx[B,S,512] ----
  const int b = bh >> 3, h = bh & 7;
#pragma unroll
  for (int r = 0; r < 8; ++r) {
    float inv = 1.0f / srun[r];
    int row = qb + r + 8 * kh;
#pragma unroll
    for (int f = 0; f < 4; ++f) {
      int col = h * 64 + f * 16 + m;
      ctx[(b * 2048 + row) * 512 + col] = bfbits(acc[f][r] * inv);
    }
  }
}

extern "C" void kernel_launch(void* const* d_in, const int* in_sizes, int n_in,
                              void* d_out, int out_size, void* d_ws, size_t ws_size,
                              hipStream_t stream) {
  const float* x  = (const float*)d_in[0];
  const float* Wq = (const float*)d_in[1];
  const float* bq = (const float*)d_in[2];
  const float* Wk = (const float*)d_in[3];
  const float* bk = (const float*)d_in[4];
  const float* Wv = (const float*)d_in[5];
  const float* bv = (const float*)d_in[6];
  const float* Wo = (const float*)d_in[7];
  const float* bo = (const float*)d_in[8];

  const size_t NE = 2u * 8u * 2048u * 64u;  // 2,097,152 elems per act tensor
  const size_t WE = 512u * 512u;            // 262,144 elems per weight
  u16* xb  = (u16*)d_ws;          // [4096,512]
  u16* WqT = xb  + NE;            // [512,512] transposed
  u16* WkT = WqT + WE;
  u16* WvT = WkT + WE;
  u16* WoT = WvT + WE;
  u16* Qb  = WoT + WE;            // [B,H,S,64]
  u16* Kb  = Qb  + NE;
  u16* VtB = Kb  + NE;            // [B,H,64,S]
  u16* ctx = VtB + NE;            // [B,S,512]

  dim3 blk(256);
  cvt_x_kernel<<<dim3(1024), blk, 0, stream>>>(x, xb);
  cvt_wt_kernel<<<dim3(128), blk, 0, stream>>>(Wq, WqT);
  cvt_wt_kernel<<<dim3(128), blk, 0, stream>>>(Wk, WkT);
  cvt_wt_kernel<<<dim3(128), blk, 0, stream>>>(Wv, WvT);
  cvt_wt_kernel<<<dim3(128), blk, 0, stream>>>(Wo, WoT);

  dim3 gg(32, 8);
  gemm512_kernel<<<gg, blk, 0, stream>>>(xb, WqT, bq, Qb,  nullptr, 0);
  gemm512_kernel<<<gg, blk, 0, stream>>>(xb, WkT, bk, Kb,  nullptr, 0);
  gemm512_kernel<<<gg, blk, 0, stream>>>(xb, WvT, bv, VtB, nullptr, 1);
  attn_kernel<<<dim3(16, 16), blk, 0, stream>>>(Qb, Kb, VtB, ctx);
  gemm512_kernel<<<gg, blk, 0, stream>>>(ctx, WoT, bo, nullptr,
                                         (float*)d_out, 2);
}
